// DTFOS_62088047231289
// MI455X (gfx1250) — matrix operand order repaired and checked by reading, not compiled, not run on hardware
//
#include <hip/hip_runtime.h>
#include <hip/hip_bf16.h>

typedef __attribute__((ext_vector_type(16))) _Float16 v16h;
typedef __attribute__((ext_vector_type(8)))  float    v8f;

namespace {
constexpr int B = 8;
constexpr int T = 8192;
constexpr int N = 128;
constexpr int TT = 256;   // conv output tile (threads per conv block)
constexpr int CH = 128;   // conv s-chunk staged in LDS
}

// ---------------------------------------------------------------------------
// Kernel 1: fractional-differencing weights, written channel-major wt[b,i,t].
// w0 = 1, w_k = w_{k-1} * (k-1-relu(alpha)) / k   (sequential cumprod).
// ---------------------------------------------------------------------------
__global__ void fd_weights_kernel(const float* __restrict__ alpha,
                                  float* __restrict__ wt) {
  const int b = blockIdx.x;       // 0..B-1
  const int i = threadIdx.x;      // 0..N-1
  float a = alpha[b * N + i];
  a = a > 0.f ? a : 0.f;          // relu
  float* wrow = wt + ((size_t)b * N + i) * T;
  float w = 1.f;
  wrow[0] = 1.f;
  for (int k = 1; k < T; ++k) {
    w *= ((float)k - 1.f - a) / (float)k;
    wrow[k] = w;
  }
}

// ---------------------------------------------------------------------------
// Kernel 2: LDS-tiled transpose X[b,t,i] -> Xt[b,i,t] (both sides coalesced).
// ---------------------------------------------------------------------------
__global__ void transpose_kernel(const float* __restrict__ X,
                                 float* __restrict__ Xt) {
  __shared__ float tile[32][33];
  const int b  = blockIdx.z;
  const int t0 = blockIdx.x * 32;
  const int i0 = blockIdx.y * 32;
#pragma unroll
  for (int r = 0; r < 4; ++r) {
    const int t = t0 + threadIdx.y + 8 * r;
    tile[threadIdx.y + 8 * r][threadIdx.x] =
        X[((size_t)b * T + t) * N + i0 + threadIdx.x];
  }
  __syncthreads();
#pragma unroll
  for (int r = 0; r < 4; ++r) {
    const int i = i0 + threadIdx.y + 8 * r;
    Xt[((size_t)b * N + i) * T + t0 + threadIdx.x] =
        tile[threadIdx.x][threadIdx.y + 8 * r];
  }
}

// ---------------------------------------------------------------------------
// Kernel 3: causal depthwise convolution (the O(T^2) fracdiff core).
// Y[t] = sum_{s=0..t} x[s] * w[t-s].  One block per (channel, 256-out tile).
// xs[c] is a wave-uniform broadcast, wsm[woff-c] is lane-consecutive ->
// conflict-free LDS reads feeding v_fmac_f32 streams.
// ---------------------------------------------------------------------------
__global__ void fd_conv_kernel(const float* __restrict__ Xt,
                               const float* __restrict__ wt,
                               float* __restrict__ Yt) {
  const int tile = blockIdx.x;         // 0..T/TT-1
  const int chan = blockIdx.y;         // 0..B*N-1
  const float* x = Xt + (size_t)chan * T;
  const float* w = wt + (size_t)chan * T;
  const int t0 = tile * TT;
  const int t  = t0 + threadIdx.x;

  __shared__ float xs[CH];
  __shared__ float wsm[TT + CH];       // covers w[t0-s0-(CH-1) .. t0+TT-1-s0]

  float acc = 0.f;
  for (int s0 = 0; s0 < t0 + TT; s0 += CH) {
    __syncthreads();
    if (threadIdx.x < CH) xs[threadIdx.x] = x[s0 + threadIdx.x];
    const int kbase = t0 - s0 - (CH - 1);
    for (int j = threadIdx.x; j < TT + CH - 1; j += TT) {
      const int k = kbase + j;
      wsm[j] = (k >= 0) ? w[k] : 0.f;
    }
    __syncthreads();
    const int lim  = min(CH, t - s0 + 1);          // enforce s <= t (causal)
    const int woff = (t - t0) + (CH - 1);
#pragma unroll 8
    for (int c = 0; c < lim; ++c) {
      acc = fmaf(xs[c], wsm[woff - c], acc);
    }
  }
  Yt[(size_t)chan * T + t] = acc;
}

// ---------------------------------------------------------------------------
// Kernel 4: fused AR residual via WMMA.
// E[b,t,j] = Y[b,t+1,j] - sum_i X[b,t,i]*A[b,j,i],  t in [0, T-2].
// One wave per 16x16 (t,j) tile; f32 accuracy via hi/lo f16 split:
// X*A ~= Xhi*Ahi + Xhi*Alo + Xlo*Ahi  -> 3x v_wmma_f32_16x16x32_f16 per
// K-chunk, fully unrolled (12 WMMAs).  Fragment data is fetched as aligned
// global_load_b128 runs (A: two 32B runs, B: two 32B runs per lane) with no
// predication, so no exec-mask manipulation around the loads.
// ---------------------------------------------------------------------------
__global__ void ar_resid_wmma_kernel(const float* __restrict__ X,
                                     const float* __restrict__ Amat,
                                     const float* __restrict__ Yt,
                                     float* __restrict__ E) {
  const int mt = blockIdx.x;           // time tile   (T/16)
  const int nt = blockIdx.y;           // j tile      (N/16)
  const int b  = blockIdx.z;
  const int lane  = threadIdx.x;       // wave32
  const int mrow  = lane & 15;
  const int hiSel = lane >> 4;         // 0: lanes 0-15, 1: lanes 16-31
  const int t0 = mt * 16;
  const int j0 = nt * 16;

  const float* Xb = X    + (size_t)b * T * N;
  const float* Ab = Amat + (size_t)b * N * N;
  // tt = t0+mrow is always < T, so loads need no guard; the last (t==T-1)
  // row's result is discarded by the store mask below.
  const float* rowX = Xb + (size_t)(t0 + mrow) * N;
  const float* rowA = Ab + (size_t)(j0 + mrow) * N;

  v8f c = {};
#pragma unroll
  for (int k0 = 0; k0 < N; k0 += 32) {
    // A fragment (16x32 f16): M = lane%16, K(h) = h + (h>=8?8:0) + hiSel*8
    const float4* pa = (const float4*)(rowX + k0 + hiSel * 8);
    const float4 a0 = pa[0];   // K rel: hiSel*8 + 0..3   (h = 0..3)
    const float4 a1 = pa[1];   // K rel: hiSel*8 + 4..7   (h = 4..7)
    const float4 a2 = pa[4];   // K rel: hiSel*8 + 16..19 (h = 8..11)
    const float4 a3 = pa[5];   // K rel: hiSel*8 + 20..23 (h = 12..15)
    // B fragment (32x16 f16): N = lane%16, K(h) = h + (h>=8?8:0) + hiSel*8
    const float4* pb = (const float4*)(rowA + k0 + hiSel * 8);
    const float4 b0 = pb[0];
    const float4 b1 = pb[1];
    const float4 b2 = pb[4];
    const float4 b3 = pb[5];

    const float af[16] = {a0.x, a0.y, a0.z, a0.w, a1.x, a1.y, a1.z, a1.w,
                          a2.x, a2.y, a2.z, a2.w, a3.x, a3.y, a3.z, a3.w};
    const float bf[16] = {b0.x, b0.y, b0.z, b0.w, b1.x, b1.y, b1.z, b1.w,
                          b2.x, b2.y, b2.z, b2.w, b3.x, b3.y, b3.z, b3.w};

    v16h ahi, alo, bhi, blo;
#pragma unroll
    for (int h = 0; h < 16; ++h) {
      const _Float16 xh = (_Float16)af[h];
      ahi[h] = xh;
      alo[h] = (_Float16)(af[h] - (float)xh);
      const _Float16 ah = (_Float16)bf[h];
      bhi[h] = ah;
      blo[h] = (_Float16)(bf[h] - (float)ah);
    }
    c = __builtin_amdgcn_wmma_f32_16x16x32_f16(false, ahi, false, bhi,
                                               (short)0, c, false, false);
    c = __builtin_amdgcn_wmma_f32_16x16x32_f16(false, ahi, false, blo,
                                               (short)0, c, false, false);
    c = __builtin_amdgcn_wmma_f32_16x16x32_f16(false, alo, false, bhi,
                                               (short)0, c, false, false);
  }

  // C/D layout: VGPR r, lane l -> M = r + (l>=16?8:0), N = l%16
  const int jn = lane & 15;
#pragma unroll
  for (int r = 0; r < 8; ++r) {
    const int t = t0 + r + (hiSel ? 8 : 0);       // output time index
    if (t < T - 1) {
      const float y = Yt[((size_t)b * N + (j0 + jn)) * T + (t + 1)];
      E[((size_t)b * (T - 1) + t) * N + j0 + jn] = y - c[r];
    }
  }
}

// ---------------------------------------------------------------------------
extern "C" void kernel_launch(void* const* d_in, const int* in_sizes, int n_in,
                              void* d_out, int out_size, void* d_ws, size_t ws_size,
                              hipStream_t stream) {
  (void)in_sizes; (void)n_in; (void)out_size; (void)ws_size;
  const float* X     = (const float*)d_in[0];   // (B, T, N)
  const float* alpha = (const float*)d_in[1];   // (B, N)
  const float* Amat  = (const float*)d_in[2];   // (B, N, N)
  float* E = (float*)d_out;                     // (B, T-1, N)

  const size_t chanTN = (size_t)B * N * T;
  float* wt = (float*)d_ws;                     // (B, N, T) weights, channel-major
  float* Xt = wt + chanTN;                      // (B, N, T) X transposed
  float* Yt = Xt + chanTN;                      // (B, N, T) fracdiff output

  fd_weights_kernel<<<dim3(B), dim3(N), 0, stream>>>(alpha, wt);
  transpose_kernel<<<dim3(T / 32, N / 32, B), dim3(32, 8), 0, stream>>>(X, Xt);
  fd_conv_kernel<<<dim3(T / TT, B * N), dim3(TT), 0, stream>>>(Xt, wt, Yt);
  ar_resid_wmma_kernel<<<dim3(T / 16, N / 16, B), dim3(32), 0, stream>>>(X, Amat, Yt, E);
}
